// Physics_Attention_1D_Eidetic_74096775790726
// MI455X (gfx1250) — compile-verified, matrix operand order checked by reading, not run on hardware
//
#include <hip/hip_runtime.h>
#include <hip/hip_bf16.h>
#include <math.h>

#define NTOK 32768
#define HH 8
#define DD 64
#define GG 64
#define CC 256
#define INNER 512
#define BB 2
#define BHN (BB*HH)   /* 16 */
#define BN (BB*NTOK)  /* 65536 */
#define KSPLIT 32
#define KT (NTOK/32)  /* 1024 k-tiles of 32 tokens per (b,h) */

typedef __attribute__((ext_vector_type(16))) __bf16 v16bf;
typedef __attribute__((ext_vector_type(8)))  __bf16 v8bf;
typedef __attribute__((ext_vector_type(8)))  float  v8f;

__device__ __forceinline__ v8f wmma_bf16(v16bf a, v16bf b, v8f c) {
  return __builtin_amdgcn_wmma_f32_16x16x32_bf16(false, a, false, b, (short)0, c, false, false);
}

// ---- fragment loaders ----
// A fragment from fp32 row-major (rows contiguous, lda floats): packed float4 loads.
__device__ __forceinline__ v16bf load_a_f32(const float* A, int lda, int k0, int lane) {
  const float* p = A + (size_t)(lane & 15) * lda + k0 + ((lane >> 4) << 3);
  const float4* q = (const float4*)p;
  float4 f0 = q[0], f1 = q[1], f2 = q[4], f3 = q[5];
  v16bf a;
  a[0]=(__bf16)f0.x; a[1]=(__bf16)f0.y; a[2]=(__bf16)f0.z; a[3]=(__bf16)f0.w;
  a[4]=(__bf16)f1.x; a[5]=(__bf16)f1.y; a[6]=(__bf16)f1.z; a[7]=(__bf16)f1.w;
  a[8]=(__bf16)f2.x; a[9]=(__bf16)f2.y; a[10]=(__bf16)f2.z; a[11]=(__bf16)f2.w;
  a[12]=(__bf16)f3.x; a[13]=(__bf16)f3.y; a[14]=(__bf16)f3.z; a[15]=(__bf16)f3.w;
  return a;
}
// A fragment from bf16 row-major: two 16B loads.
__device__ __forceinline__ v16bf load_a_bf(const __bf16* A, int lda, int k0, int lane) {
  const __bf16* p = A + (size_t)(lane & 15) * lda + k0 + ((lane >> 4) << 3);
  v8bf lo = *(const v8bf*)p;
  v8bf hi = *(const v8bf*)(p + 16);
  v16bf a;
  #pragma unroll
  for (int j = 0; j < 8; ++j) { a[j] = lo[j]; a[8 + j] = hi[j]; }
  return a;
}
// B fragment from fp32 row-major [K x ldb] (strided scalar; one-time conversions only).
__device__ __forceinline__ v16bf load_b_f32(const float* B, int ldb, int k0, int col0, int lane) {
  const float* p = B + (size_t)k0 * ldb + col0 + (lane & 15);
  int kb = (lane >> 4) << 3;
  v16bf b;
  #pragma unroll
  for (int j = 0; j < 16; ++j) {
    int k = kb + ((j < 8) ? j : (j + 8));
    b[j] = (__bf16)p[(size_t)k * ldb];
  }
  return b;
}

__device__ __forceinline__ float gelu_f(float v) {
  return 0.5f * v * (1.0f + erff(v * 0.70710678118654752f));
}

// -------- kprep: one-time conversion of weights into fragment-swizzled bf16 --------
__global__ void kprep(const float* __restrict__ Wx, const float* __restrict__ W1,
                      const float* __restrict__ Ws, const float* __restrict__ Wo,
                      __bf16* __restrict__ wxs, __bf16* __restrict__ w1s,
                      __bf16* __restrict__ wss, __bf16* __restrict__ wos) {
  int f = blockIdx.x, lane = threadIdx.x;
  const float* src; int ldb, kt, ct; __bf16* dst; int fi;
  if (f < 256)      { fi = f;       src = Wx; ldb = INNER; kt = fi / 32; ct = fi % 32; dst = wxs; }
  else if (f < 512) { fi = f - 256; src = Wo; ldb = CC;    kt = fi / 16; ct = fi % 16; dst = wos; }
  else if (f < 520) { fi = f - 512; src = W1; ldb = GG;    kt = fi / 4;  ct = fi % 4;  dst = w1s; }
  else              { fi = f - 520; src = Ws; ldb = GG;    kt = fi / 4;  ct = fi % 4;  dst = wss; }
  v16bf b = load_b_f32(src, ldb, kt * 32, ct * 16, lane);
  ((v16bf*)dst)[(size_t)fi * 32 + lane] = b;
}

// -------- K1: x_mid = reshape(x@Wx+bx)+rope. Wave = 16 rows x 128 cols (8 accs). --------
__global__ void k1_xproj(const float* __restrict__ x, const float* __restrict__ pos,
                         const __bf16* __restrict__ wxs, const float* __restrict__ bx,
                         const float* __restrict__ Wr, const float* __restrict__ br,
                         float* __restrict__ xmid, __bf16* __restrict__ xmB) {
  int lane = threadIdx.x;
  int mt = blockIdx.x;            // BN/16 row tiles
  int nt0 = blockIdx.y * 8;       // 8 col tiles per wave
  const float* A = x + (size_t)mt * 16 * CC;
  const v16bf* Bf = (const v16bf*)wxs;
  v8f acc[8] = {};
  for (int kt = 0; kt < CC / 32; ++kt) {
    v16bf a = load_a_f32(A, CC, kt * 32, lane);
    #pragma unroll
    for (int j = 0; j < 8; ++j) {
      v16bf b = Bf[((size_t)kt * (INNER / 16) + nt0 + j) * 32 + lane];
      acc[j] = wmma_bf16(a, b, acc[j]);
    }
  }
  int rbase = (lane >> 4) << 3;
  int m0 = mt * 16;
  int b_ = m0 >> 15;
  int t0 = m0 & (NTOK - 1);
  // hoist per-row sin/cos out of the column loop
  float ss[8][3], cs[8][3];
  #pragma unroll
  for (int r = 0; r < 8; ++r) {
    int t = t0 + r + rbase;
    #pragma unroll
    for (int p = 0; p < 3; ++p) {
      float invf = __expf(-(float)p * 3.070113457f);  // ln(10000)/3
      float ang = pos[((size_t)b_ * NTOK + t) * 3 + p] * invf;
      ss[r][p] = __sinf(ang); cs[r][p] = __cosf(ang);
    }
  }
  #pragma unroll
  for (int j = 0; j < 8; ++j) {
    int nt = nt0 + j;
    int n = nt * 16 + (lane & 15);
    int h = nt >> 2, d = n & 63;
    int bh = b_ * HH + h;
    float bxi = bx[n], brd = br[d];
    float wr_s[3], wr_c[3];
    #pragma unroll
    for (int p = 0; p < 3; ++p) { wr_s[p] = Wr[p * DD + d]; wr_c[p] = Wr[(3 + p) * DD + d]; }
    float val[8];
    #pragma unroll
    for (int r = 0; r < 8; ++r) {
      int t = t0 + r + rbase;
      float rope = brd;
      #pragma unroll
      for (int p = 0; p < 3; ++p) rope += ss[r][p] * wr_s[p] + cs[r][p] * wr_c[p];
      val[r] = acc[j][r] + bxi + rope;
      xmid[((size_t)bh * NTOK + t) * DD + d] = val[r];
    }
    v8bf pk;
    #pragma unroll
    for (int r = 0; r < 8; ++r) pk[r] = (__bf16)val[r];
    size_t frag = ((size_t)bh * KT + (t0 >> 5)) * 4 + (nt & 3);
    *(v8bf*)(xmB + frag * 512 + (size_t)lane * 16 + ((t0 >> 4) & 1) * 8) = pk;
  }
}

// -------- K2: temperature + gumbel softmax; writes bf16 sw (row-major) + swT fragments ------
__global__ void k2_sliceweights(const float* __restrict__ xmid, const float* __restrict__ gum,
                                const __bf16* __restrict__ w1s, const float* __restrict__ b1,
                                const float* __restrict__ W2, const float* __restrict__ b2,
                                const float* __restrict__ bias,
                                const __bf16* __restrict__ wss, const float* __restrict__ bs,
                                __bf16* __restrict__ swB, __bf16* __restrict__ swT) {
  __shared__ float t1s[16][GG];
  __shared__ float lgs[16][GG];
  int lane = threadIdx.x;
  int n0 = blockIdx.x * 16;
  int bh = blockIdx.y;
  const float* A = xmid + ((size_t)bh * NTOK + n0) * DD;
  const v16bf* B1 = (const v16bf*)w1s;
  const v16bf* Bs = (const v16bf*)wss;
  v16bf a0 = load_a_f32(A, DD, 0, lane);
  v16bf a1 = load_a_f32(A, DD, 32, lane);
  int col = lane & 15;
  int rbase = (lane >> 4) << 3;
  #pragma unroll
  for (int ct = 0; ct < 4; ++ct) {
    v8f acc = {};
    acc = wmma_bf16(a0, B1[(0 * 4 + ct) * 32 + lane], acc);
    acc = wmma_bf16(a1, B1[(1 * 4 + ct) * 32 + lane], acc);
    #pragma unroll
    for (int r = 0; r < 8; ++r) t1s[r + rbase][ct * 16 + col] = acc[r];
    v8f acc2 = {};
    acc2 = wmma_bf16(a0, Bs[(0 * 4 + ct) * 32 + lane], acc2);
    acc2 = wmma_bf16(a1, Bs[(1 * 4 + ct) * 32 + lane], acc2);
    #pragma unroll
    for (int r = 0; r < 8; ++r) lgs[r + rbase][ct * 16 + col] = acc2[r];
  }
  __syncthreads();
  if (lane < 16) {
    int row = lane;
    int h = bh & (HH - 1);
    float u = 0.f;
    for (int g = 0; g < GG; ++g) u += gelu_f(t1s[row][g] + b1[g]) * W2[g];
    float temp = fmaxf(gelu_f(u + b2[0]) + bias[h], 0.01f);
    float inv_t = 1.0f / temp;
    size_t base = ((size_t)bh * NTOK + n0 + row) * GG;
    float mx = -1e30f;
    for (int g0 = 0; g0 < GG; g0 += 4) {
      float4 u4 = *(const float4*)(gum + base + g0);
      float uu[4] = {u4.x, u4.y, u4.z, u4.w};
      #pragma unroll
      for (int q = 0; q < 4; ++q) {
        float gn = -__logf(-__logf(uu[q] + 1e-8f) + 1e-8f);
        float z = (lgs[row][g0 + q] + bs[g0 + q] + gn) * inv_t;
        lgs[row][g0 + q] = z;
        mx = fmaxf(mx, z);
      }
    }
    float s = 0.f;
    for (int g = 0; g < GG; ++g) { float e = __expf(lgs[row][g] - mx); lgs[row][g] = e; s += e; }
    float inv = 1.0f / s;
    for (int g0 = 0; g0 < GG; g0 += 8) {
      v8bf pk;
      #pragma unroll
      for (int j = 0; j < 8; ++j) {
        float v = lgs[row][g0 + j] * inv;
        lgs[row][g0 + j] = v;
        pk[j] = (__bf16)v;
      }
      *(v8bf*)(swB + base + g0) = pk;
    }
  }
  __syncthreads();
  // cooperative write of A-side (sw^T) fragments for k3
  int half = (n0 >> 4) & 1;
  #pragma unroll
  for (int mt = 0; mt < 4; ++mt) {
    int g = mt * 16 + (lane & 15);
    v8bf pk;
    #pragma unroll
    for (int j = 0; j < 8; ++j) pk[j] = (__bf16)lgs[rbase + j][g];
    size_t frag = ((size_t)bh * KT + (n0 >> 5)) * 4 + mt;
    *(v8bf*)(swT + frag * 512 + (size_t)lane * 16 + half * 8) = pk;
  }
}

// -------- K2b: snorm[b,h,g] = sum_n sw[b,h,n,g] --------
__global__ void k2b_snorm(const __bf16* __restrict__ swB, float* __restrict__ snorm) {
  __shared__ float red[256];
  int idx = blockIdx.x;           // bh*64 + g
  int bh = idx >> 6, g = idx & 63;
  const __bf16* p = swB + (size_t)bh * NTOK * GG + g;
  float s = 0.f;
  for (int n = threadIdx.x; n < NTOK; n += 256) s += (float)p[(size_t)n * GG];
  red[threadIdx.x] = s;
  __syncthreads();
  for (int off = 128; off > 0; off >>= 1) {
    if ((int)threadIdx.x < off) red[threadIdx.x] += red[threadIdx.x + off];
    __syncthreads();
  }
  if (threadIdx.x == 0) snorm[idx] = red[0];
}

// -------- K3: st_raw = sw^T @ x_mid over N; one wave owns the full 64x64 tile --------
__global__ void k3_pool(const __bf16* __restrict__ swT, const __bf16* __restrict__ xmB,
                        float* __restrict__ straw) {
  int lane = threadIdx.x;
  int bh = blockIdx.x / KSPLIT;
  int kc = blockIdx.x % KSPLIT;
  const v16bf* Af = (const v16bf*)swT;
  const v16bf* Bf = (const v16bf*)xmB;
  v8f acc[4][4] = {};
  int kt0 = kc * (KT / KSPLIT);
  for (int kt = kt0; kt < kt0 + KT / KSPLIT; ++kt) {
    v16bf a[4], b[4];
    size_t base = ((size_t)bh * KT + kt) * 4;
    #pragma unroll
    for (int mt = 0; mt < 4; ++mt) a[mt] = Af[(base + mt) * 32 + lane];
    #pragma unroll
    for (int ct = 0; ct < 4; ++ct) b[ct] = Bf[(base + ct) * 32 + lane];
    #pragma unroll
    for (int mt = 0; mt < 4; ++mt)
      #pragma unroll
      for (int ct = 0; ct < 4; ++ct)
        acc[mt][ct] = wmma_bf16(a[mt], b[ct], acc[mt][ct]);
  }
  int rbase = (lane >> 4) << 3;
  #pragma unroll
  for (int mt = 0; mt < 4; ++mt)
    #pragma unroll
    for (int ct = 0; ct < 4; ++ct) {
      int colc = ct * 16 + (lane & 15);
      #pragma unroll
      for (int r = 0; r < 8; ++r) {
        int m = mt * 16 + r + rbase;
        atomicAdd(&straw[(size_t)bh * GG * DD + (size_t)m * DD + colc], acc[mt][ct][r]);
      }
    }
}

// -------- K4: tiny per-(b,h) 64x64 SDPA + sigmoid gate --------
__global__ void k4_sdpa(const float* __restrict__ straw, const float* __restrict__ snorm,
                        const float* __restrict__ Wq, const float* __restrict__ Wk,
                        const float* __restrict__ Wv,
                        const float* __restrict__ Wg1, const float* __restrict__ bg1,
                        const float* __restrict__ Wg2, const float* __restrict__ bg2,
                        float* __restrict__ outst) {
  __shared__ float st[GG][DD];
  __shared__ float kk[GG][DD];
  __shared__ float vv[GG][DD];
  int g = threadIdx.x;            // 0..63
  int bh = blockIdx.x;
  const float* sr = straw + (size_t)bh * GG * DD + (size_t)g * DD;
  float inv = 1.0f / (snorm[bh * GG + g] + 1e-5f);
  for (int d = 0; d < DD; ++d) st[g][d] = sr[d] * inv;
  __syncthreads();
  float q[DD];
  for (int d = 0; d < DD; ++d) {
    float aq = 0.f, ak = 0.f, av = 0.f;
    for (int c = 0; c < DD; ++c) {
      float s = st[g][c];
      aq += s * Wq[c * DD + d];
      ak += s * Wk[c * DD + d];
      av += s * Wv[c * DD + d];
    }
    q[d] = aq; kk[g][d] = ak; vv[g][d] = av;
  }
  __syncthreads();
  float attn[GG];
  float mx = -1e30f;
  for (int j = 0; j < GG; ++j) {
    float s = 0.f;
    for (int d = 0; d < DD; ++d) s += q[d] * kk[j][d];
    s *= 0.125f;
    attn[j] = s;
    mx = fmaxf(mx, s);
  }
  float ssum = 0.f;
  for (int j = 0; j < GG; ++j) { attn[j] = __expf(attn[j] - mx); ssum += attn[j]; }
  float isum = 1.0f / ssum;
  float out[DD];
  for (int d = 0; d < DD; ++d) {
    float s = 0.f;
    for (int j = 0; j < GG; ++j) s += attn[j] * vv[j][d];
    out[d] = s * isum;
  }
  float y[DD];
  for (int e = 0; e < DD; ++e) {
    float s = bg1[e];
    for (int c = 0; c < DD; ++c) s += st[g][c] * Wg1[c * DD + e];
    for (int c = 0; c < DD; ++c) s += out[c] * Wg1[(DD + c) * DD + e];
    y[e] = s / (1.0f + __expf(-s));   // silu
  }
  float* op = outst + (size_t)bh * GG * DD + (size_t)g * DD;
  for (int d = 0; d < DD; ++d) {
    float s = bg2[d];
    for (int e = 0; e < DD; ++e) s += y[e] * Wg2[e * DD + d];
    float gate = 1.0f / (1.0f + __expf(-s));
    op[d] = gate * out[d];
  }
}

// -------- K4b: swizzle out_st into bf16 B-fragments for k5 --------
__global__ void k4b_prep(const float* __restrict__ outst, __bf16* __restrict__ outstB) {
  int bh = blockIdx.x, lane = threadIdx.x;
  const float* S = outst + (size_t)bh * GG * DD;
  #pragma unroll
  for (int f = 0; f < 8; ++f) {
    int kt = f >> 2, ct = f & 3;
    v16bf b = load_b_f32(S, DD, kt * 32, ct * 16, lane);
    ((v16bf*)outstB)[((size_t)bh * 8 + f) * 32 + lane] = b;
  }
}

// -------- K5: out_x = sw @ out_st. Wave = 16 rows x 64 cols (4 accs). --------
__global__ void k5_broadcast(const __bf16* __restrict__ swB, const __bf16* __restrict__ outstB,
                             float* __restrict__ outx) {
  int lane = threadIdx.x;
  int mt = blockIdx.x;            // NTOK/16
  int bh = blockIdx.y;
  const __bf16* A = swB + ((size_t)bh * NTOK + (size_t)mt * 16) * GG;
  const v16bf* Bf = (const v16bf*)outstB + (size_t)bh * 8 * 32;
  v8f acc[4] = {};
  #pragma unroll
  for (int kt = 0; kt < 2; ++kt) {
    v16bf a = load_a_bf(A, GG, kt * 32, lane);
    #pragma unroll
    for (int ct = 0; ct < 4; ++ct)
      acc[ct] = wmma_bf16(a, Bf[(kt * 4 + ct) * 32 + lane], acc[ct]);
  }
  int rbase = (lane >> 4) << 3;
  float* O = outx + ((size_t)bh * NTOK + (size_t)mt * 16) * DD;
  #pragma unroll
  for (int ct = 0; ct < 4; ++ct) {
    int colc = ct * 16 + (lane & 15);
    #pragma unroll
    for (int r = 0; r < 8; ++r) O[(size_t)(r + rbase) * DD + colc] = acc[ct][r];
  }
}

// -------- K6: out = out_x(reshaped) @ Wo + bo. Wave = 16 rows x 256 cols (16 accs). --------
__global__ void k6_outproj(const float* __restrict__ outx, const __bf16* __restrict__ wos,
                           const float* __restrict__ bo, float* __restrict__ out) {
  int lane = threadIdx.x;
  int mt = blockIdx.x;            // BN/16
  const v16bf* Bf = (const v16bf*)wos;
  v8f acc[16] = {};
  int m = mt * 16 + (lane & 15);
  int b_ = m >> 15;
  int t = m & (NTOK - 1);
  int kb = (lane >> 4) << 3;
  for (int kt = 0; kt < INNER / 32; ++kt) {
    int kA = kt * 32 + kb;
    v16bf a;
    {
      int h0 = kA >> 6, d0 = kA & 63;
      const float* pr = outx + (((size_t)b_ * HH + h0) * NTOK + t) * DD + d0;
      float4 f0 = *(const float4*)pr, f1 = *(const float4*)(pr + 4);
      a[0]=(__bf16)f0.x; a[1]=(__bf16)f0.y; a[2]=(__bf16)f0.z; a[3]=(__bf16)f0.w;
      a[4]=(__bf16)f1.x; a[5]=(__bf16)f1.y; a[6]=(__bf16)f1.z; a[7]=(__bf16)f1.w;
    }
    {
      int k2i = kA + 16;
      int h0 = k2i >> 6, d0 = k2i & 63;
      const float* pr = outx + (((size_t)b_ * HH + h0) * NTOK + t) * DD + d0;
      float4 f0 = *(const float4*)pr, f1 = *(const float4*)(pr + 4);
      a[8]=(__bf16)f0.x; a[9]=(__bf16)f0.y; a[10]=(__bf16)f0.z; a[11]=(__bf16)f0.w;
      a[12]=(__bf16)f1.x; a[13]=(__bf16)f1.y; a[14]=(__bf16)f1.z; a[15]=(__bf16)f1.w;
    }
    #pragma unroll
    for (int nt = 0; nt < 16; ++nt) {
      v16bf b = Bf[((size_t)kt * (CC / 16) + nt) * 32 + lane];
      acc[nt] = wmma_bf16(a, b, acc[nt]);
    }
  }
  #pragma unroll
  for (int nt = 0; nt < 16; ++nt) {
    int colc = nt * 16 + (lane & 15);
    float bon = bo[colc];
    #pragma unroll
    for (int r = 0; r < 8; ++r)
      out[(size_t)(mt * 16 + r + kb) * CC + colc] = acc[nt][r] + bon;
  }
}

extern "C" void kernel_launch(void* const* d_in, const int* in_sizes, int n_in,
                              void* d_out, int out_size, void* d_ws, size_t ws_size,
                              hipStream_t stream) {
  (void)in_sizes; (void)n_in; (void)out_size; (void)ws_size;
  const float* x    = (const float*)d_in[0];
  const float* pos  = (const float*)d_in[1];
  const float* gum  = (const float*)d_in[2];
  const float* Wx   = (const float*)d_in[3];
  const float* bx   = (const float*)d_in[4];
  const float* W1   = (const float*)d_in[5];
  const float* b1   = (const float*)d_in[6];
  const float* W2   = (const float*)d_in[7];
  const float* b2   = (const float*)d_in[8];
  const float* bias = (const float*)d_in[9];
  const float* Wr   = (const float*)d_in[10];
  const float* br   = (const float*)d_in[11];
  const float* Ws   = (const float*)d_in[12];
  const float* bs   = (const float*)d_in[13];
  const float* Wq   = (const float*)d_in[14];
  const float* Wk   = (const float*)d_in[15];
  const float* Wv   = (const float*)d_in[16];
  const float* Wg1  = (const float*)d_in[17];
  const float* bg1  = (const float*)d_in[18];
  const float* Wg2  = (const float*)d_in[19];
  const float* bg2  = (const float*)d_in[20];
  const float* Wo   = (const float*)d_in[21];
  const float* bo   = (const float*)d_in[22];
  float* out = (float*)d_out;

  char* ws = (char*)d_ws;
  const size_t BIGF = (size_t)BN * INNER;              // 33,554,432 elements
  float*  xmid   = (float*)ws;                          ws += BIGF * 4;   // 134MB (aliased outx)
  __bf16* xmB    = (__bf16*)ws;                         ws += BIGF * 2;   // 67MB B-frag swizzled
  __bf16* swB    = (__bf16*)ws;                         ws += BIGF * 2;   // 67MB row-major bf16
  __bf16* swT    = (__bf16*)ws;                         ws += BIGF * 2;   // 67MB A-frag swizzled
  float*  straw  = (float*)ws;                          ws += (size_t)BHN * GG * DD * 4;
  float*  snorm  = (float*)ws;                          ws += (size_t)BHN * GG * 4;
  float*  outst  = (float*)ws;                          ws += (size_t)BHN * GG * DD * 4;
  __bf16* outstB = (__bf16*)ws;                         ws += (size_t)BHN * GG * DD * 2;
  __bf16* wxs    = (__bf16*)ws;                         ws += (size_t)CC * INNER * 2;
  __bf16* wos    = (__bf16*)ws;                         ws += (size_t)INNER * CC * 2;
  __bf16* w1s    = (__bf16*)ws;                         ws += (size_t)DD * GG * 2;
  __bf16* wss    = (__bf16*)ws;                         ws += (size_t)DD * GG * 2;
  float*  outx   = xmid;   // fp32 x_mid is dead after k2 -> reuse for out_x

  kprep<<<dim3(528), 32, 0, stream>>>(Wx, W1, Ws, Wo, wxs, w1s, wss, wos);
  k1_xproj<<<dim3(BN / 16, INNER / 128), 32, 0, stream>>>(x, pos, wxs, bx, Wr, br, xmid, xmB);
  k2_sliceweights<<<dim3(NTOK / 16, BHN), 32, 0, stream>>>(xmid, gum, w1s, b1, W2, b2, bias,
                                                           wss, bs, swB, swT);
  k2b_snorm<<<dim3(BHN * GG), 256, 0, stream>>>(swB, snorm);
  hipMemsetAsync(straw, 0, (size_t)BHN * GG * DD * sizeof(float), stream);
  k3_pool<<<dim3(BHN * KSPLIT), 32, 0, stream>>>(swT, xmB, straw);
  k4_sdpa<<<dim3(BHN), 64, 0, stream>>>(straw, snorm, Wq, Wk, Wv, Wg1, bg1, Wg2, bg2, outst);
  k4b_prep<<<dim3(BHN), 32, 0, stream>>>(outst, outstB);
  k5_broadcast<<<dim3(NTOK / 16, BHN), 32, 0, stream>>>(swB, outstB, outx);
  k6_outproj<<<dim3(BN / 16), 32, 0, stream>>>(outx, wos, bo, out);
}
